// GRU_56547539419564
// MI455X (gfx1250) — matrix operand order repaired and checked by reading, not compiled, not run on hardware
//
#include <hip/hip_runtime.h>
#include <hip/hip_bf16.h>

// ---------------------------------------------------------------------------
// Bidirectional GRU + fuser GRU for MI455X (gfx1250, wave32, WMMA).
// B=64, T=512, H=512.  All matmuls via v_wmma_f32_16x16x32_bf16 (f32 accum).
// ---------------------------------------------------------------------------

#define Bv 64
#define Tv 512
#define Hv 512
#define Gv 1536           // 3H
#define TBv 32768         // T*B
#define HP 520            // padded LDS row stride (bf16 elems) -> conflict-free b128

typedef __attribute__((ext_vector_type(8)))  float  v8f;
typedef __attribute__((ext_vector_type(16))) __bf16 v16bf;
typedef __attribute__((ext_vector_type(8)))  __bf16 v8bf;

union V16 { v16bf v; v8bf h[2]; };

__device__ __forceinline__ v8f vzero8() {
  v8f z = {0.f,0.f,0.f,0.f,0.f,0.f,0.f,0.f};
  return z;
}

__device__ __forceinline__ v8f wmma_bf16(V16 a, V16 b, v8f c) {
  return __builtin_amdgcn_wmma_f32_16x16x32_bf16(
      /*neg_a=*/false, a.v, /*neg_b=*/false, b.v,
      /*c_mod=*/(short)0, c, /*reuse_a=*/false, /*reuse_b=*/false);
}

// ---------------------------------------------------------------------------
// f32 -> bf16 cast (weights + x), grid-stride
// ---------------------------------------------------------------------------
__global__ void cast_f32_bf16(const float* __restrict__ in,
                              __bf16* __restrict__ out, int n) {
  int i = blockIdx.x * blockDim.x + threadIdx.x;
  int stride = gridDim.x * blockDim.x;
  for (; i < n; i += stride) out[i] = (__bf16)in[i];
}

// ---------------------------------------------------------------------------
// gx GEMM:  out[m, 0..1536) = A_row(m) . W^T + bias        (M=32768, N=1536)
//   xmode==1: A row for m=(t*64+b) is x[b*T + t] (lda=K=512)
//   xmode==0: A row m directly (cat, lda=K=1024)
// Block = 256 threads (8 waves); each wave owns 1 M-tile x 12 N-tiles.
// ---------------------------------------------------------------------------
__global__ __launch_bounds__(256)
void gemm_gx(const __bf16* __restrict__ A, const __bf16* __restrict__ W,
             const float* __restrict__ bias, float* __restrict__ out,
             int K, int xmode) {
  const int wave = threadIdx.x >> 5;
  const int lane = threadIdx.x & 31;
  const int half = lane >> 4;
  const int ln   = lane & 15;
  const int mtile = blockIdx.x;

  const int m = mtile * 16 + ln;   // A-row supplied by this lane
  const size_t arow = xmode ? ((size_t)(m & 63) * Tv + (size_t)(m >> 6))
                            : (size_t)m;
  const __bf16* Arow = A + arow * (size_t)K;

  v8f acc[12];
  #pragma unroll
  for (int i = 0; i < 12; ++i) acc[i] = vzero8();

  const int nkb = K / 32;
  for (int kb = 0; kb < nkb; ++kb) {
    const __bf16* ap = Arow + kb * 32 + 8 * half;
    V16 af; af.h[0] = *(const v8bf*)ap; af.h[1] = *(const v8bf*)(ap + 16);
    #pragma unroll
    for (int nt = 0; nt < 12; ++nt) {
      const int c = (wave * 12 + nt) * 16 + ln;          // W row (= out col)
      const __bf16* wp = W + (size_t)c * K + kb * 32 + 8 * half;
      V16 bfv; bfv.h[0] = *(const v8bf*)wp; bfv.h[1] = *(const v8bf*)(wp + 16);
      acc[nt] = wmma_bf16(af, bfv, acc[nt]);
    }
  }

  #pragma unroll
  for (int nt = 0; nt < 12; ++nt) {
    const int c = (wave * 12 + nt) * 16 + ln;
    const float bv = bias[c];
    #pragma unroll
    for (int rr = 0; rr < 8; ++rr) {
      const int mg = mtile * 16 + rr + 8 * half;
      out[(size_t)mg * Gv + c] = acc[nt][rr] + bv;
    }
  }
}

// ---------------------------------------------------------------------------
// GRU scan: one block = one full scan (1024 thr = 32 waves on one WGP).
//   h (64x512): f32 in registers (owner lanes), bf16 in LDS for WMMA A-frags.
//   Per step: gh = h . whh^T (64x1536x512 via 6144 WMMAs across 32 waves),
//             gate math, write h' (LDS bf16 + global).
// grid=2 runs fwd & bwd scans concurrently.
// ---------------------------------------------------------------------------
struct ScanJob {
  const float*  gx;     // (T,B,3H) f32, bih already folded in
  const __bf16* whh;    // (3H,H) bf16 row-major
  const float*  bhh;    // (3H) f32
  float*        outf;   // mode 1: (B,T,H) f32
  __bf16*       outb;   // mode 0: cat base (+col offset), row stride 2H
  int reverse;
  int mode;             // 0: bf16->cat, 1: f32->out
};

__global__ __launch_bounds__(1024)
void gru_scan(ScanJob j0, ScanJob j1) {
  const ScanJob j = (blockIdx.x == 0) ? j0 : j1;
  extern __shared__ __bf16 hbf[];          // 64 rows x HP bf16 (66,560 B)

  const int tid  = threadIdx.x;
  const int wave = tid >> 5;
  const int lane = tid & 31;
  const int half = lane >> 4;
  const int ln   = lane & 15;
  const int mt   = wave >> 3;              // 0..3  (b-row tile)
  const int ntg0 = (wave & 7) * 4;         // 4 column tiles per wave

  for (int i = tid; i < Bv * HP; i += 1024) hbf[i] = (__bf16)0.0f;
  __syncthreads();

  // per-lane constants: output columns + recurrent biases
  int   colA[4];
  float br[4], bz[4], bn[4];
  #pragma unroll
  for (int a = 0; a < 4; ++a) {
    const int col = (ntg0 + a) * 16 + ln;
    colA[a] = col;
    br[a] = j.bhh[col];
    bz[a] = j.bhh[Hv + col];
    bn[a] = j.bhh[2 * Hv + col];
  }

  float hreg[4][8];
  #pragma unroll
  for (int a = 0; a < 4; ++a)
    #pragma unroll
    for (int r = 0; r < 8; ++r) hreg[a][r] = 0.f;

  for (int step = 0; step < Tv; ++step) {
    const int t = j.reverse ? (Tv - 1 - step) : step;

    v8f acc[3][4];
    #pragma unroll
    for (int g = 0; g < 3; ++g)
      #pragma unroll
      for (int a = 0; a < 4; ++a) acc[g][a] = vzero8();

    // ---- gh = h_bf16 . whh^T -------------------------------------------
    for (int kb = 0; kb < Hv / 32; ++kb) {
      const int rowb = mt * 16 + ln;
      const __bf16* hp = &hbf[rowb * HP + kb * 32 + 8 * half];
      V16 af; af.h[0] = *(const v8bf*)hp; af.h[1] = *(const v8bf*)(hp + 16);
      #pragma unroll
      for (int a = 0; a < 4; ++a) {
        const int c0 = (ntg0 + a) * 16 + ln;           // column within H
        #pragma unroll
        for (int g = 0; g < 3; ++g) {
          const __bf16* wp = j.whh + (size_t)(g * Hv + c0) * Hv
                                   + kb * 32 + 8 * half;
          V16 bfv; bfv.h[0] = *(const v8bf*)wp; bfv.h[1] = *(const v8bf*)(wp + 16);
          acc[g][a] = wmma_bf16(af, bfv, acc[g][a]);
        }
      }
    }

    // ---- gates + state update ------------------------------------------
    #pragma unroll
    for (int a = 0; a < 4; ++a) {
      const int i = colA[a];
      #pragma unroll
      for (int rr = 0; rr < 8; ++rr) {
        const int b = mt * 16 + rr + 8 * half;
        const float* gxr = j.gx + ((size_t)t * Bv + b) * Gv;
        const float xr = gxr[i];
        const float xz = gxr[Hv + i];
        const float xn = gxr[2 * Hv + i];
        const float hr = acc[0][a][rr] + br[a];
        const float hz = acc[1][a][rr] + bz[a];
        const float hn = acc[2][a][rr] + bn[a];
        const float rg = 1.f / (1.f + __expf(-(xr + hr)));
        const float zg = 1.f / (1.f + __expf(-(xz + hz)));
        const float ng = tanhf(xn + rg * hn);
        hreg[a][rr] = (1.f - zg) * ng + zg * hreg[a][rr];
      }
    }

    __syncthreads();   // everyone done reading old hbf

    #pragma unroll
    for (int a = 0; a < 4; ++a) {
      const int i = colA[a];
      #pragma unroll
      for (int rr = 0; rr < 8; ++rr) {
        const int b = mt * 16 + rr + 8 * half;
        const float hv = hreg[a][rr];
        hbf[b * HP + i] = (__bf16)hv;
        if (j.mode == 0)
          j.outb[((size_t)t * Bv + b) * (2 * Hv) + i] = (__bf16)hv;
        else
          j.outf[((size_t)b * Tv + t) * Hv + i] = hv;
      }
    }

    __syncthreads();   // hbf updated for next step
  }
}

// ---------------------------------------------------------------------------
extern "C" void kernel_launch(void* const* d_in, const int* in_sizes, int n_in,
                              void* d_out, int out_size, void* d_ws, size_t ws_size,
                              hipStream_t stream) {
  (void)in_sizes; (void)n_in; (void)out_size; (void)ws_size;

  const float* x     = (const float*)d_in[0];
  const float* wih_f = (const float*)d_in[1];
  const float* whh_f = (const float*)d_in[2];
  const float* bih_f = (const float*)d_in[3];
  const float* bhh_f = (const float*)d_in[4];
  const float* wih_b = (const float*)d_in[5];
  const float* whh_b = (const float*)d_in[6];
  const float* bih_b = (const float*)d_in[7];
  const float* bhh_b = (const float*)d_in[8];
  const float* wih_u = (const float*)d_in[9];
  const float* whh_u = (const float*)d_in[10];
  const float* bih_u = (const float*)d_in[11];
  const float* bhh_u = (const float*)d_in[12];
  float* out = (float*)d_out;

  char* ws = (char*)d_ws;
  size_t off = 0;
  auto wsalloc = [&](size_t bytes) -> char* {
    char* p = ws + off;
    off += (bytes + 255) & ~(size_t)255;
    return p;
  };

  __bf16* xbf  = (__bf16*)wsalloc((size_t)Bv * Tv * Hv * 2);     // 32 MB
  __bf16* wfI  = (__bf16*)wsalloc((size_t)Gv * Hv * 2);          // wih_f
  __bf16* wfH  = (__bf16*)wsalloc((size_t)Gv * Hv * 2);          // whh_f
  __bf16* wbI  = (__bf16*)wsalloc((size_t)Gv * Hv * 2);          // wih_b
  __bf16* wbH  = (__bf16*)wsalloc((size_t)Gv * Hv * 2);          // whh_b
  __bf16* wuI  = (__bf16*)wsalloc((size_t)Gv * 2 * Hv * 2);      // wih_u
  __bf16* wuH  = (__bf16*)wsalloc((size_t)Gv * Hv * 2);          // whh_u
  float*  gxf  = (float*)wsalloc((size_t)TBv * Gv * 4);          // 192 MB
  float*  gxb  = (float*)wsalloc((size_t)TBv * Gv * 4);          // 192 MB
  float*  gxu  = (float*)wsalloc((size_t)TBv * Gv * 4);          // 192 MB
  __bf16* cat  = (__bf16*)wsalloc((size_t)TBv * 2 * Hv * 2);     // 64 MB

  // ---- stage 0: bf16 casts -------------------------------------------------
  cast_f32_bf16<<<4096, 256, 0, stream>>>(x, xbf, Bv * Tv * Hv);
  cast_f32_bf16<<<1024, 256, 0, stream>>>(wih_f, wfI, Gv * Hv);
  cast_f32_bf16<<<1024, 256, 0, stream>>>(whh_f, wfH, Gv * Hv);
  cast_f32_bf16<<<1024, 256, 0, stream>>>(wih_b, wbI, Gv * Hv);
  cast_f32_bf16<<<1024, 256, 0, stream>>>(whh_b, wbH, Gv * Hv);
  cast_f32_bf16<<<2048, 256, 0, stream>>>(wih_u, wuI, Gv * 2 * Hv);
  cast_f32_bf16<<<1024, 256, 0, stream>>>(whh_u, wuH, Gv * Hv);

  // ---- stage 1: input-gate GEMMs (parallel work, WMMA) --------------------
  gemm_gx<<<TBv / 16, 256, 0, stream>>>(xbf, wfI, bih_f, gxf, Hv, 1);
  gemm_gx<<<TBv / 16, 256, 0, stream>>>(xbf, wbI, bih_b, gxb, Hv, 1);

  // ---- stage 2: fwd + bwd scans concurrently (2 blocks, 1 WGP each) -------
  ScanJob jf; jf.gx = gxf; jf.whh = wfH; jf.bhh = bhh_f;
  jf.outf = nullptr; jf.outb = cat;        jf.reverse = 0; jf.mode = 0;
  ScanJob jb; jb.gx = gxb; jb.whh = wbH; jb.bhh = bhh_b;
  jb.outf = nullptr; jb.outb = cat + Hv;   jb.reverse = 1; jb.mode = 0;
  gru_scan<<<2, 1024, Bv * HP * 2, stream>>>(jf, jb);

  // ---- stage 3: fuser input GEMM over cat (K=2H) --------------------------
  gemm_gx<<<TBv / 16, 256, 0, stream>>>(cat, wuI, bih_u, gxu, 2 * Hv, 0);

  // ---- stage 4: fuser scan, writes (B,T,H) f32 output ---------------------
  ScanJob ju; ju.gx = gxu; ju.whh = wuH; ju.bhh = bhh_u;
  ju.outf = out; ju.outb = nullptr; ju.reverse = 0; ju.mode = 1;
  gru_scan<<<1, 1024, Bv * HP * 2, stream>>>(ju, ju);
}
